// HetGnnEncoder_84361747628644
// MI455X (gfx1250) — matrix operand order repaired and checked by reading, not compiled, not run on hardware
//
#include <hip/hip_runtime.h>

#define BT 16384   // nodes
#define KF 3       // fanout groups
#define FN 10      // neighbors
#define DD 128     // feature dim
#define HH 64      // hidden per direction
#define TT 11      // sequence length (node + 10 neighbors)

typedef __attribute__((ext_vector_type(16))) _Float16 v16h;
typedef __attribute__((ext_vector_type(8)))  _Float16 v8h;
typedef __attribute__((ext_vector_type(8)))  float    v8f;
typedef __attribute__((ext_vector_type(4)))  float    v4f;

__device__ __forceinline__ float sigf(float x) { return 1.f / (1.f + __expf(-x)); }

// Build a WMMA A-operand (16x32 f16) from an LDS tile.
// Lane L holds row M=L%16; elements 0..15 = K in
// [koff + (L<16?0:16), +16) (contiguous, matching the B operand).
__device__ __forceinline__ v16h load_a_tile(const _Float16* base, int row, int ld,
                                            int koff, int lane) {
  const _Float16* p = base + row * ld + koff + ((lane < 16) ? 0 : 16);
  v8h lo = *(const v8h*)p;
  v8h hi = *(const v8h*)(p + 8);
  v16h a;
#pragma unroll
  for (int i = 0; i < 8; ++i) { a[i] = lo[i]; a[i + 8] = hi[i]; }
  return a;
}

__global__ __launch_bounds__(256) void convert_weights_kernel(
    const float* __restrict__ wih, const float* __restrict__ whh,
    _Float16* __restrict__ wih_h, _Float16* __restrict__ whh_h) {
  int i = blockIdx.x * 256 + threadIdx.x;
  if (i < KF * 2 * 4 * HH * DD) wih_h[i] = (_Float16)wih[i];
  if (i < KF * 2 * 4 * HH * HH) whh_h[i] = (_Float16)whh[i];
}

// One workgroup = 16 batch rows for one (fanout k, direction). 8 waves, each
// wave owns 2 of the 16 gate N-tiles (256 gate outputs). Gate GEMMs via WMMA
// f16->f32; LSTM state kept in f32 in LDS; h re-fed as f16.
__global__ __launch_bounds__(256) void bilstm_kernel(
    const float* __restrict__ node, const float* __restrict__ neigh,
    const _Float16* __restrict__ wih_h, const _Float16* __restrict__ whh_h,
    const float* __restrict__ lstm_b, float* __restrict__ embeds) {
  constexpr int XLD = DD + 8;  // padded LDS row strides (bank-conflict avoidance,
  constexpr int HLD = HH + 8;  // still 16B aligned for v8h loads)
  __shared__ __align__(16) _Float16 x_lds[16 * XLD];
  __shared__ __align__(16) _Float16 h_lds[16 * HLD];
  __shared__ float gates_lds[16 * 256];
  __shared__ float c_lds[16 * HH];
  __shared__ float hsum_lds[16 * HH];

  const int tid = threadIdx.x;
  const int wave = tid >> 5, lane = tid & 31;
  const int kd = blockIdx.y;           // k*2 + dir
  const int k = kd >> 1, dir = kd & 1;
  const int b0 = blockIdx.x * 16;

  // init state
#pragma unroll
  for (int q = 0; q < 4; ++q) {
    int idx = tid + q * 256;           // 0..1023 == 16 rows x 64 hidden
    c_lds[idx] = 0.f;
    hsum_lds[idx] = 0.f;
    int m = idx >> 6, j = idx & 63;
    h_lds[m * HLD + j] = (_Float16)0.f;
  }

  // Preload gate weights into registers in WMMA-B layout.
  // B operand 32x16 f16: lane L holds column N=L%16, elements e = K of
  // (L<16?0:16)+e within the k-block -> 16 *contiguous* f16 of W[n, :].
  const int n0 = wave * 32;            // this wave's two N-tiles: n0, n0+16
  const int ncol = lane & 15;
  const int khi = (lane < 16) ? 0 : 16;
  v16h Bw1[4][2];                      // x @ Wih^T : K=128 -> 4 k-blocks of 32
  v16h Bw2[2][2];                      // h @ Whh^T : K=64  -> 2 k-blocks of 32
#pragma unroll
  for (int nt = 0; nt < 2; ++nt) {
    const _Float16* w1 = wih_h + (size_t)(kd * 256 + n0 + nt * 16 + ncol) * DD + khi;
#pragma unroll
    for (int kb = 0; kb < 4; ++kb) Bw1[kb][nt] = *(const v16h*)(w1 + kb * 32);
    const _Float16* w2 = whh_h + (size_t)(kd * 256 + n0 + nt * 16 + ncol) * HH + khi;
#pragma unroll
    for (int kb = 0; kb < 2; ++kb) Bw2[kb][nt] = *(const v16h*)(w2 + kb * 32);
  }
  const float bias0 = lstm_b[kd * 256 + n0 + ncol];
  const float bias1 = lstm_b[kd * 256 + n0 + 16 + ncol];

  const int xm = tid >> 4;             // staging: row 0..15
  const int xc = (tid & 15) * 8;       // staging: 8 features per thread
  const int mrow = lane & 15;
  const int mbase = (lane < 16) ? 0 : 8;

  for (int tt = 0; tt < TT; ++tt) {
    const int ts = dir ? (TT - 1 - tt) : tt;  // backward dir walks seq reversed
    // ---- stage x[t] tile: f32 global -> f16 LDS ----
    const float* xsrc =
        (ts == 0) ? (node + (size_t)(b0 + xm) * DD + xc)
                  : (neigh + (((size_t)k * BT + (b0 + xm)) * FN + (ts - 1)) * DD + xc);
    v4f f0 = *(const v4f*)xsrc;
    v4f f1 = *(const v4f*)(xsrc + 4);
    v8h xv;
#pragma unroll
    for (int i = 0; i < 4; ++i) { xv[i] = (_Float16)f0[i]; xv[4 + i] = (_Float16)f1[i]; }
    *(v8h*)(x_lds + xm * XLD + xc) = xv;
    if (tt + 1 < TT) {  // prefetch next timestep's tile (global_prefetch_b8)
      const int tn = dir ? (TT - 2 - tt) : (tt + 1);
      const float* nx =
          (tn == 0) ? (node + (size_t)(b0 + xm) * DD + xc)
                    : (neigh + (((size_t)k * BT + (b0 + xm)) * FN + (tn - 1)) * DD + xc);
      __builtin_prefetch(nx, 0, 0);
    }
    __syncthreads();

    // ---- gates = x @ Wih^T + h @ Whh^T + b   (WMMA f16 -> f32 acc) ----
    v8f acc0 = {}, acc1 = {};
#pragma unroll
    for (int kb = 0; kb < 4; ++kb) {
      v16h a = load_a_tile(x_lds, mrow, XLD, kb * 32, lane);
      acc0 = __builtin_amdgcn_wmma_f32_16x16x32_f16(false, a, false, Bw1[kb][0],
                                                    (short)0, acc0, false, false);
      acc1 = __builtin_amdgcn_wmma_f32_16x16x32_f16(false, a, false, Bw1[kb][1],
                                                    (short)0, acc1, false, false);
    }
#pragma unroll
    for (int kb = 0; kb < 2; ++kb) {
      v16h a = load_a_tile(h_lds, mrow, HLD, kb * 32, lane);
      acc0 = __builtin_amdgcn_wmma_f32_16x16x32_f16(false, a, false, Bw2[kb][0],
                                                    (short)0, acc0, false, false);
      acc1 = __builtin_amdgcn_wmma_f32_16x16x32_f16(false, a, false, Bw2[kb][1],
                                                    (short)0, acc1, false, false);
    }
    // C/D layout: VGPR r, lanes 0-15 -> (M=r, N=lane), lanes 16-31 -> (M=r+8, N=lane-16)
#pragma unroll
    for (int r = 0; r < 8; ++r) {
      gates_lds[(mbase + r) * 256 + n0 + ncol] = acc0[r] + bias0;
      gates_lds[(mbase + r) * 256 + n0 + 16 + ncol] = acc1[r] + bias1;
    }
    __syncthreads();

    // ---- pointwise LSTM cell update (f32), gate order i,f,g,o ----
#pragma unroll
    for (int q = 0; q < 4; ++q) {
      int idx = tid + q * 256;
      int m = idx >> 6, j = idx & 63;
      const float* g = gates_lds + m * 256;
      float gi = g[j], gf = g[64 + j], gg = g[128 + j], go = g[192 + j];
      float c = c_lds[idx];
      float cn = sigf(gf) * c + sigf(gi) * tanhf(gg);
      float h = sigf(go) * tanhf(cn);
      c_lds[idx] = cn;
      hsum_lds[idx] += h;                 // time-mean is order independent
      h_lds[m * HLD + j] = (_Float16)h;   // feed back to next step's WMMA
    }
    __syncthreads();
  }

  // ---- write mean hidden states into embeds[k][b][dir*64 + j] ----
#pragma unroll
  for (int q = 0; q < 4; ++q) {
    int idx = tid + q * 256;
    int m = idx >> 6, j = idx & 63;
    embeds[((size_t)k * BT + (b0 + m)) * DD + dir * HH + j] =
        hsum_lds[idx] * (1.f / (float)TT);
  }
}

// One wave32 per node: 4-source attention (3 fanout embeds + node) + mix.
__global__ __launch_bounds__(256) void attn_kernel(
    const float* __restrict__ node, const float* __restrict__ embeds,
    const float* __restrict__ att_w, const float* __restrict__ att_b,
    float* __restrict__ out) {
  const int wave = threadIdx.x >> 5, lane = threadIdx.x & 31;
  const size_t b = (size_t)blockIdx.x * 8 + wave;
  const int d0 = lane * 4;
  v4f nd = *(const v4f*)(node + b * DD + d0);
  v4f wn = *(const v4f*)(att_w + d0);
  v4f we = *(const v4f*)(att_w + DD + d0);
  v4f e0 = *(const v4f*)(embeds + ((size_t)0 * BT + b) * DD + d0);
  v4f e1 = *(const v4f*)(embeds + ((size_t)1 * BT + b) * DD + d0);
  v4f e2 = *(const v4f*)(embeds + ((size_t)2 * BT + b) * DD + d0);
  float s0 = 0, s1 = 0, s2 = 0, s3 = 0, s4 = 0;
#pragma unroll
  for (int i = 0; i < 4; ++i) {
    s0 += wn[i] * nd[i];   // att_w[:128] . node   (shared term)
    s1 += we[i] * e0[i];
    s2 += we[i] * e1[i];
    s3 += we[i] * e2[i];
    s4 += we[i] * nd[i];   // node acting as 4th source
  }
#pragma unroll
  for (int off = 16; off > 0; off >>= 1) {
    s0 += __shfl_xor(s0, off, 32);
    s1 += __shfl_xor(s1, off, 32);
    s2 += __shfl_xor(s2, off, 32);
    s3 += __shfl_xor(s3, off, 32);
    s4 += __shfl_xor(s4, off, 32);
  }
  const float bb = att_b[0];
  float l0 = s0 + s1 + bb; l0 = (l0 > 0.f) ? l0 : 0.01f * l0;
  float l1 = s0 + s2 + bb; l1 = (l1 > 0.f) ? l1 : 0.01f * l1;
  float l2 = s0 + s3 + bb; l2 = (l2 > 0.f) ? l2 : 0.01f * l2;
  float l3 = s0 + s4 + bb; l3 = (l3 > 0.f) ? l3 : 0.01f * l3;
  float mx = fmaxf(fmaxf(l0, l1), fmaxf(l2, l3));
  float x0 = __expf(l0 - mx), x1 = __expf(l1 - mx);
  float x2 = __expf(l2 - mx), x3 = __expf(l3 - mx);
  float inv = 1.f / (x0 + x1 + x2 + x3);
  v4f o;
#pragma unroll
  for (int i = 0; i < 4; ++i)
    o[i] = (x0 * e0[i] + x1 * e1[i] + x2 * e2[i] + x3 * nd[i]) * inv;
  *(v4f*)(out + b * DD + d0) = o;
}

extern "C" void kernel_launch(void* const* d_in, const int* in_sizes, int n_in,
                              void* d_out, int out_size, void* d_ws, size_t ws_size,
                              hipStream_t stream) {
  (void)in_sizes; (void)n_in; (void)out_size; (void)ws_size;
  const float* node   = (const float*)d_in[0];  // [B,128]
  const float* neigh  = (const float*)d_in[1];  // [3,B,10,128]
  const float* wih    = (const float*)d_in[2];  // [3,2,256,128]
  const float* whh    = (const float*)d_in[3];  // [3,2,256,64]
  const float* lstm_b = (const float*)d_in[4];  // [3,2,256]
  const float* att_w  = (const float*)d_in[5];  // [1,256]
  const float* att_b  = (const float*)d_in[6];  // [1]
  float* out = (float*)d_out;                   // [B,128]

  char* ws = (char*)d_ws;
  _Float16* wih_h = (_Float16*)ws;                       // 393216 B
  _Float16* whh_h = (_Float16*)(ws + 393216);            // 196608 B
  float* embeds   = (float*)(ws + 589824);               // [3,B,128] f32, 25 MB

  convert_weights_kernel<<<768, 256, 0, stream>>>(wih, whh, wih_h, whh_h);
  bilstm_kernel<<<dim3(BT / 16, KF * 2), 256, 0, stream>>>(node, neigh, wih_h,
                                                           whh_h, lstm_b, embeds);
  attn_kernel<<<BT / 8, 256, 0, stream>>>(node, embeds, att_w, att_b, out);
}
